// RQKMeans_4140348473632
// MI455X (gfx1250) — matrix operand order repaired and checked by reading, not compiled, not run on hardware
//
#include <hip/hip_runtime.h>

#define N_LAYERS 4
#define K_CB 1024
#define F 512
#define TILE_M 128
#define NT 64        // 1024/16 codeword tiles
#define KC 16        // 512/32 k-chunks
#define BTILE_BYTES (KC * 32 * 16 * 2)   // 16 KB per B tile
#define BTILE_ELEMS (KC * 32 * 16)       // 8192 bf16 elements

typedef __bf16 bf16_t;
typedef bf16_t v16bf __attribute__((ext_vector_type(16)));
typedef float v8f __attribute__((ext_vector_type(8)));
typedef unsigned short u16;
typedef unsigned int u32;
typedef u32 u32x4 __attribute__((ext_vector_type(4)));
typedef u32 v4u __attribute__((ext_vector_type(4)));
typedef int v8i __attribute__((ext_vector_type(8)));
typedef int v4i __attribute__((ext_vector_type(4)));

__device__ __forceinline__ u16 f2bf(float f) {
    u32 u = __float_as_uint(f);
    u32 r = u + 0x7FFFu + ((u >> 16) & 1u);   // round-to-nearest-even
    return (u16)(r >> 16);
}
__device__ __forceinline__ float bf2f(u16 h) {
    return __uint_as_float(((u32)h) << 16);
}

// ---------------------------------------------------------------------------
// TDM: DMA one contiguous 16 KB bf16 tile from global into LDS.
// D# per CDNA5 ISA ch.8: group0 = {count=1 | lds_addr | global_addr | type=2},
// group1 = {data_size=2B, tensor_dim0=tile_dim0=8192, dim1=1, stride0=8192}.
// Issued by one wave; completion tracked with TENSORcnt.
// ---------------------------------------------------------------------------
#if __has_builtin(__builtin_amdgcn_tensor_load_to_lds)
#define HAVE_TDM 1
__device__ __forceinline__ void tdm_load_btile(const u16* gsrc, u16* ldst) {
    unsigned long long ga = (unsigned long long)(uintptr_t)gsrc;
    u32 lds = (u32)(uintptr_t)ldst;           // low 32 bits of generic ptr = LDS offset
    v4u g0;
    g0.x = 0x1u;                                             // count=1, user descriptor
    g0.y = lds;                                              // lds_addr
    g0.z = (u32)(ga & 0xFFFFFFFFu);                          // global_addr[31:0]
    g0.w = (u32)((ga >> 32) & 0x01FFFFFFu) | (2u << 30);     // global_addr[56:32] | type=2
    v8i g1;
    g1[0] = (1 << 16);                                       // data_size=1 (2 bytes)
    g1[1] = (int)((BTILE_ELEMS & 0xFFFF) << 16);             // tensor_dim0[15:0]
    g1[2] = (BTILE_ELEMS >> 16) | (1 << 16);                 // tensor_dim0[31:16] | tensor_dim1 lo
    g1[3] = (int)((BTILE_ELEMS & 0xFFFF) << 16);             // tensor_dim1 hi=0 | tile_dim0
    g1[4] = 1;                                               // tile_dim1=1, tile_dim2=0
    g1[5] = BTILE_ELEMS;                                     // tensor_dim0_stride[31:0]
    g1[6] = 0;
    g1[7] = 0;
    v4i z = {0, 0, 0, 0};
#if __clang_major__ >= 23
    v8i z8 = {0, 0, 0, 0, 0, 0, 0, 0};
    __builtin_amdgcn_tensor_load_to_lds(g0, g1, z, z, z8, 0);
#else
    __builtin_amdgcn_tensor_load_to_lds(g0, g1, z, z, 0);
#endif
}
#else
#define HAVE_TDM 0
#endif

// ---------------------------------------------------------------------------
// Prep 1: pack codebooks (f32 [L][K][F]) into bf16 B-fragment layout.
// ---------------------------------------------------------------------------
__global__ __launch_bounds__(256) void pack_cb(const float* __restrict__ cb,
                                               u16* __restrict__ cbB) {
    int b   = blockIdx.x;          // l*1024 + nt*16 + kch
    int kch = b & 15;
    int nt  = (b >> 4) & 63;
    int l   = b >> 10;
    const float* base = cb + (size_t)l * K_CB * F;
    u16* out = cbB + (size_t)b * 512;
    for (int i = threadIdx.x; i < 512; i += 256) {
        int ln  = i >> 4;
        int e   = i & 15;
        int col = (nt << 4) + (ln & 15);
        int k   = (kch << 5) + (e & 7) + ((ln >> 4) << 3) + ((e >> 3) << 4);
        out[i] = f2bf(base[(size_t)col * F + k]);
    }
}

// ---------------------------------------------------------------------------
// Prep 2: c2[l][k] = sum(cb[l][k]^2) in f32 (exact).
// ---------------------------------------------------------------------------
__global__ __launch_bounds__(256) void c2_kernel(const float* __restrict__ cb,
                                                 float* __restrict__ c2g) {
    __shared__ float red[256];
    int row = blockIdx.x;
    const float* p = cb + (size_t)row * F;
    float s = 0.f;
    for (int i = threadIdx.x; i < F; i += 256) { float v = p[i]; s += v * v; }
    red[threadIdx.x] = s;
    __syncthreads();
    for (int o = 128; o > 0; o >>= 1) {
        if (threadIdx.x < (unsigned)o) red[threadIdx.x] += red[threadIdx.x + o];
        __syncthreads();
    }
    if (threadIdx.x == 0) c2g[row] = red[0];
}

// ---------------------------------------------------------------------------
// Fused RQ encode+decode, TDM double-buffered B staging.
// ---------------------------------------------------------------------------
__global__ __launch_bounds__(256) void rq_main(const float* __restrict__ data,
                                               const float* __restrict__ cb,
                                               const u16* __restrict__ cbB,
                                               const float* __restrict__ c2g,
                                               long long* __restrict__ codes,
                                               float* __restrict__ recon,
                                               int n) {
    __shared__ __attribute__((aligned(32))) u16 resA[TILE_M * F];      // 128 KB
    __shared__ __attribute__((aligned(32))) u16 sB[2][BTILE_ELEMS];    // 2 x 16 KB
    __shared__ float sc2[K_CB];                                        // 4 KB
    __shared__ int rowIdx[TILE_M];

    const int tid  = threadIdx.x;
    const int w    = tid >> 5;
    const int lane = tid & 31;
    const long long rowBase = (long long)blockIdx.x * TILE_M;

    // ---- Phase 0: stream data tile (coalesced f32x4), pack bf16 A-fragments
    for (int i = tid; i < TILE_M * F / 4; i += 256) {
        int e4 = i << 2;
        int r  = e4 >> 9;
        int k  = e4 & (F - 1);
        long long grow = rowBase + r;
        if (grow >= n) grow = n - 1;
        const float4 v = ((const float4*)(data + grow * F))[k >> 2];
        float vv[4] = {v.x, v.y, v.z, v.w};
#pragma unroll
        for (int j = 0; j < 4; ++j) {
            int kk  = k + j;
            int kch = kk >> 5;
            int kl  = kk & 31;
            int ln  = (r & 15) + (((kl >> 3) & 1) << 4);
            int e   = (kl & 7) + ((kl >> 4) << 3);
            resA[(((r >> 4) * KC + kch) * 32 + ln) * 16 + e] = f2bf(vv[j]);
        }
    }

    for (int l = 0; l < N_LAYERS; ++l) {
        const u16* cbBl = cbB + (size_t)l * NT * BTILE_ELEMS;
        for (int i = tid; i < K_CB; i += 256) sc2[i] = c2g[l * K_CB + i];

#if HAVE_TDM
        if (w == 0) tdm_load_btile(cbBl, &sB[0][0]);     // prefetch tile 0
#endif

        float bd[8];
        int   bi[8];
#pragma unroll
        for (int j = 0; j < 8; ++j) { bd[j] = 3.4e38f; bi[j] = 0; }

        for (int nt = 0; nt < NT; ++nt) {
#if HAVE_TDM
            if (w == 0) __builtin_amdgcn_s_wait_tensorcnt(0);  // tile nt landed in LDS
            __syncthreads();                                   // publish; all done with other buf
            if (w == 0 && nt + 1 < NT)                         // DMA next tile during compute
                tdm_load_btile(cbBl + (size_t)(nt + 1) * BTILE_ELEMS, &sB[(nt + 1) & 1][0]);
            const u16* bbuf = &sB[nt & 1][0];
#else
            __syncthreads();
            {   // synchronous fallback staging
                const u32x4* src = (const u32x4*)(cbBl + (size_t)nt * BTILE_ELEMS);
                u32x4* dst = (u32x4*)&sB[0][0];
                for (int i = tid; i < BTILE_ELEMS / 8; i += 256) dst[i] = src[i];
            }
            __syncthreads();
            const u16* bbuf = &sB[0][0];
#endif
            v8f acc = {0.f, 0.f, 0.f, 0.f, 0.f, 0.f, 0.f, 0.f};
#pragma unroll
            for (int c = 0; c < KC; ++c) {
                v16bf a = *(const v16bf*)&resA[((w * KC + c) * 32 + lane) * 16];
                v16bf b = *(const v16bf*)&bbuf[(c * 32 + lane) * 16];
                acc = __builtin_amdgcn_wmma_f32_16x16x32_bf16(
                    false, a, false, b, (short)0, acc, false, false);
            }
            int col   = (nt << 4) + (lane & 15);
            float c2v = sc2[col];
#pragma unroll
            for (int j = 0; j < 8; ++j) {           // D layout: vgpr j -> M = j + 8*(lane/16)
                float s = c2v - 2.0f * acc[j];
                if (s < bd[j]) { bd[j] = s; bi[j] = col; }
            }
#if !HAVE_TDM
            __syncthreads();
#endif
        }

        // ---- finalize per-row argmin: min across the 16 lanes of each half
        const float* cbl = cb + (size_t)l * K_CB * F;
#pragma unroll
        for (int j = 0; j < 8; ++j) {
            float d = bd[j];
            int idx = bi[j];
#pragma unroll
            for (int m = 1; m < 16; m <<= 1) {
                float od = __shfl_xor(d, m, 32);
                int   oi = __shfl_xor(idx, m, 32);
                if (od < d || (od == d && oi < idx)) { d = od; idx = oi; }
            }
            if ((lane & 15) == 0) {
                int row = (w << 4) + j + ((lane >> 4) << 3);
                rowIdx[row] = idx;
                long long grow = rowBase + row;
                if (grow < n) codes[grow * N_LAYERS + l] = (long long)idx;
            }
        }
        __syncthreads();

        // ---- residual update: r -= cb[idx] in f32 (exact gather from f32 codebook)
        for (int c = 0; c < KC; ++c) {
            int idx   = rowIdx[(w << 4) + (lane & 15)];
            int kbase = (c << 5) + ((lane >> 4) << 3);
            const float* q = cbl + (size_t)idx * F + kbase;
            float4 q0 = ((const float4*)q)[0];
            float4 q1 = ((const float4*)q)[1];
            float4 q2 = ((const float4*)(q + 16))[0];
            float4 q3 = ((const float4*)(q + 16))[1];
            float qq[16] = {q0.x, q0.y, q0.z, q0.w, q1.x, q1.y, q1.z, q1.w,
                            q2.x, q2.y, q2.z, q2.w, q3.x, q3.y, q3.z, q3.w};
            u16* p = &resA[((w * KC + c) * 32 + lane) * 16];
#pragma unroll
            for (int e = 0; e < 16; ++e) p[e] = f2bf(bf2f(p[e]) - qq[e]);
        }
        __syncthreads();
    }

    // ---- recon = data - residual_final (coalesced read + write)
    for (int i = tid; i < TILE_M * F / 4; i += 256) {
        int e4 = i << 2;
        int r  = e4 >> 9;
        int k  = e4 & (F - 1);
        long long grow = rowBase + r;
        if (grow >= n) continue;
        const float4 v = ((const float4*)(data + grow * F))[k >> 2];
        float vv[4] = {v.x, v.y, v.z, v.w};
        float out[4];
#pragma unroll
        for (int j = 0; j < 4; ++j) {
            int kk  = k + j;
            int kch = kk >> 5;
            int kl  = kk & 31;
            int ln  = (r & 15) + (((kl >> 3) & 1) << 4);
            int e   = (kl & 7) + ((kl >> 4) << 3);
            out[j] = vv[j] - bf2f(resA[(((r >> 4) * KC + kch) * 32 + ln) * 16 + e]);
        }
        float4 o4 = {out[0], out[1], out[2], out[3]};
        ((float4*)(recon + grow * F))[k >> 2] = o4;
    }
}

// ---------------------------------------------------------------------------
extern "C" void kernel_launch(void* const* d_in, const int* in_sizes, int n_in,
                              void* d_out, int out_size, void* d_ws, size_t ws_size,
                              hipStream_t stream) {
    const float* data = (const float*)d_in[0];
    const float* cb   = (const float*)d_in[1];
    int n = in_sizes[0] / F;   // 100000

    u16*   cbB = (u16*)d_ws;                                              // 4 MiB packed bf16
    float* c2g = (float*)((char*)d_ws + (size_t)N_LAYERS * K_CB * F * sizeof(u16));

    long long* codes = (long long*)d_out;                 // [n][4] int64 first
    float*     recon = (float*)(codes + (size_t)n * N_LAYERS);

    pack_cb<<<N_LAYERS * NT * KC, 256, 0, stream>>>(cb, cbB);
    c2_kernel<<<N_LAYERS * K_CB, 256, 0, stream>>>(cb, c2g);
    int blocks = (n + TILE_M - 1) / TILE_M;
    rq_main<<<blocks, 256, 0, stream>>>(data, cb, cbB, c2g, codes, recon, n);
}
